// Model_Recursive_LSTM_v2_37168646979654
// MI455X (gfx1250) — matrix operand order repaired and checked by reading, not compiled, not run on hardware
//
#include <hip/hip_runtime.h>
#include <math.h>

// ---------------- model constants ----------------
#define BB      256
#define CC      32
#define BC      8192      // B*C
#define TEXPR   64
#define TTRANS  16
#define NLOOPS  12
#define MAXTAGS 50
#define LOOPSF  8

// ---------------- workspace layout (float offsets) ----------------
#define WS_WCF   0                      // folded fwd input weights 128x64 (K padded 50->64, zeroed)
#define WS_BF    (WS_WCF + 128*64)      // folded fwd bias 128
#define WS_WCB   (WS_BF  + 128)         // folded bwd input weights 128x64
#define WS_BB    (WS_WCB + 128*64)      // folded bwd bias 128
#define WS_PROG  (WS_BB  + 128)         // prog (8192 x 64): fwd h | bwd h
#define WS_EXPR  (WS_PROG + BC*64)      // expr emb (8192 x 16), cols 11..15 zero
#define WS_CEMB  (WS_EXPR + BC*16)      // comps_emb (8192 x 16)

typedef __attribute__((ext_vector_type(16))) _Float16 v16h;
typedef __attribute__((ext_vector_type(8)))  float    v8f;
typedef int v4i_g __attribute__((vector_size(16)));   // matches builtin param type

__device__ __forceinline__ float sigf(float x) { return 1.0f / (1.0f + __expf(-x)); }
__device__ __forceinline__ float eluf(float x) { return x > 0.0f ? x : (__expf(x) - 1.0f); }

// ---------------- async global->LDS copies (CDNA5, ASYNCcnt) ----------------
#if defined(__gfx1250__) && __has_builtin(__builtin_amdgcn_global_load_async_to_lds_b32) && \
    __has_builtin(__builtin_amdgcn_global_load_async_to_lds_b128) && \
    __has_builtin(__builtin_amdgcn_s_wait_asynccnt)
#define USE_ASYNC 1
#else
#define USE_ASYNC 0
#endif

// AS(1) = global ("__device__"), AS(3) = LDS; low 32 bits of generic LDS addr = LDS offset
#define AS1_I32(p) ((__attribute__((address_space(1))) int*)(unsigned long long)(p))
#define AS3_I32(p) ((__attribute__((address_space(3))) int*)(unsigned)(unsigned long long)(p))
#define AS1_V4I(p) ((__attribute__((address_space(1))) v4i_g*)(unsigned long long)(p))
#define AS3_V4I(p) ((__attribute__((address_space(3))) v4i_g*)(unsigned)(unsigned long long)(p))

__device__ __forceinline__ void cp_b32(const float* g, float* l) {
#if USE_ASYNC
  __builtin_amdgcn_global_load_async_to_lds_b32(AS1_I32(g), AS3_I32(l), 0, 0);
#else
  *l = *g;
#endif
}
__device__ __forceinline__ void cp_b128(const float* g, float* l) {
#if USE_ASYNC
  __builtin_amdgcn_global_load_async_to_lds_b128(AS1_V4I(g), AS3_V4I(l), 0, 0);
#else
  l[0] = g[0]; l[1] = g[1]; l[2] = g[2]; l[3] = g[3];
#endif
}
__device__ __forceinline__ void async_wait() {
#if USE_ASYNC
  __builtin_amdgcn_s_wait_asynccnt(0);
#endif
}

// ---- WMMA 16x16x32 f16 helpers (wave32 layouts per CDNA5 ISA 7.12.2) ----
// A (16x32 f16): lane L holds row M=L&15; element j -> K = (j&7) + (j>=8?16:0) + (L>=16?8:0)
template <class F>
__device__ __forceinline__ v16h load_A(F f, int lane) {
  v16h a;
  const int m  = lane & 15;
  const int ko = (lane & 16) ? 8 : 0;
#pragma unroll
  for (int j = 0; j < 16; ++j) {
    int k = (j & 7) + ((j & 8) ? 16 : 0) + ko;
    a[j] = (_Float16)f(m, k);
  }
  return a;
}
// B (32x16 f16): lane L holds col N=L&15; element j -> K = (j&7) + (j>=8?16:0) + (L>=16?8:0)
template <class F>
__device__ __forceinline__ v16h load_B(F f, int lane) {
  v16h b;
  const int n  = lane & 15;
  const int ko = (lane & 16) ? 8 : 0;
#pragma unroll
  for (int j = 0; j < 16; ++j) b[j] = (_Float16)f((j & 7) + ((j & 8) ? 16 : 0) + ko, n);
  return b;
}
__device__ __forceinline__ v8f wmma16(v16h a, v16h b, v8f c) {
  return __builtin_amdgcn_wmma_f32_16x16x32_f16(false, a, false, b, (short)0, c, false, false);
}
__device__ __forceinline__ v8f acc_bias(float bv) {
  v8f a;
#pragma unroll
  for (int r = 0; r < 8; ++r) a[r] = bv;
  return a;
}
// D (16x16 f32): lane L col N=L&15, element r -> row M = r + (L>=16?8:0)
template <class S>
__device__ __forceinline__ void store_D(v8f d, int lane, S s) {
  const int n  = lane & 15;
  const int mo = (lane & 16) ? 8 : 0;
#pragma unroll
  for (int r = 0; r < 8; ++r) s(mo + r, n, d[r]);
}

// ================= K0: fold encode into trans-LSTM input path =================
__global__ void __launch_bounds__(256)
prep_kernel(const float* __restrict__ We,   const float* __restrict__ be,
            const float* __restrict__ WihF, const float* __restrict__ bihF,
            const float* __restrict__ bhhF,
            const float* __restrict__ WihB, const float* __restrict__ bihB,
            const float* __restrict__ bhhB,
            float* __restrict__ ws) {
  const int t   = threadIdx.x;      // 0..255
  const int dir = t >> 7;
  const int g   = t & 127;
  const float* Wih = dir ? WihB : WihF;
  float* wc   = ws + (dir ? WS_WCB : WS_WCF);
  float* bias = ws + (dir ? WS_BB  : WS_BF);
  for (int j = 0; j < 64; ++j) {
    float s = 0.0f;
    if (j < MAXTAGS)
      for (int n = 0; n < MAXTAGS; ++n) s += Wih[g * MAXTAGS + n] * We[n * MAXTAGS + j];
    wc[g * 64 + j] = s;
  }
  float s = dir ? (bihB[g] + bhhB[g]) : (bihF[g] + bhhF[g]);
  for (int n = 0; n < MAXTAGS; ++n) s += Wih[g * MAXTAGS + n] * be[n];
  bias[g] = s;
}

// ================= K1: exprs_lstm (in 11, hidden 11, seq 64, batch 8192) =====
__global__ void __launch_bounds__(32)
expr_lstm_kernel(const float* __restrict__ expr,
                 const float* __restrict__ Wih, const float* __restrict__ Whh,
                 const float* __restrict__ bih, const float* __restrict__ bhh,
                 float* __restrict__ ws) {
  __shared__ float xb[2][16 * 16];   // double-buffered x tile (padded cols 11..15 = 0)
  __shared__ float z[16 * 64];
  __shared__ float h[16 * 16];
  const int lane = threadIdx.x;
  const int row0 = blockIdx.x * 16;
  float* exprOut = ws + WS_EXPR;

  for (int i = lane; i < 256; i += 32) { h[i] = 0.0f; xb[0][i] = 0.0f; xb[1][i] = 0.0f; }
  float creg[6];
#pragma unroll
  for (int s = 0; s < 6; ++s) creg[s] = 0.0f;

  v16h bt[4];
  float bias[4];
#pragma unroll
  for (int nt = 0; nt < 4; ++nt) {
    bt[nt] = load_B([&](int k, int n) -> float {
      int ng = nt * 16 + n;
      if (ng >= 44) return 0.0f;
      if (k < 16) return (k < 11) ? Wih[ng * 11 + k] : 0.0f;
      int kk = k - 16;
      return (kk < 11) ? Whh[ng * 11 + kk] : 0.0f;
    }, lane);
    int ng = nt * 16 + (lane & 15);
    bias[nt] = (ng < 44) ? (bih[ng] + bhh[ng]) : 0.0f;
  }

  auto stage = [&](int t, int buf) {   // 16 rows x 11 floats, coalesced async
#pragma unroll
    for (int s = 0; s < 6; ++s) {
      int p = lane + 32 * s;
      if (p < 176) {
        int m = p / 11, k = p % 11;
        cp_b32(expr + (size_t)(row0 + m) * (TEXPR * 11) + t * 11 + k, &xb[buf][m * 16 + k]);
      }
    }
  };

  stage(0, 0);
  async_wait();
  __syncthreads();

  for (int t = 0; t < TEXPR; ++t) {
    const int cur = t & 1;
    if (t + 1 < TEXPR) stage(t + 1, cur ^ 1);   // prefetch next step while computing
    v16h a = load_A([&](int m, int k) -> float {
      if (k < 16) return xb[cur][m * 16 + k];    // compile-time per element
      return h[m * 16 + (k - 16)];
    }, lane);
#pragma unroll
    for (int nt = 0; nt < 4; ++nt) {
      v8f d = acc_bias(bias[nt]);
      d = wmma16(a, bt[nt], d);
      store_D(d, lane, [&](int m, int n, float v) { z[m * 64 + nt * 16 + n] = v; });
    }
    __syncthreads();
#pragma unroll
    for (int s = 0; s < 6; ++s) {
      int p = lane + 32 * s;
      if (p < 176) {                       // 16 rows x 11 hidden
        int m = p / 11, j = p % 11;
        float zi = z[m * 64 + j],      zf = z[m * 64 + 11 + j];
        float zg = z[m * 64 + 22 + j], zo = z[m * 64 + 33 + j];
        float c = sigf(zf) * creg[s] + sigf(zi) * tanhf(zg);
        creg[s] = c;
        h[m * 16 + j] = sigf(zo) * tanhf(c);
      }
    }
    async_wait();
    __syncthreads();
  }
  for (int i = lane; i < 256; i += 32)
    exprOut[(size_t)(row0 + i / 16) * 16 + (i % 16)] = h[i];  // cols 11..15 stay 0
}

// ================= K2: trans fwd/bwd LSTM (in 50 folded, hidden 32) ==========
__global__ void __launch_bounds__(32)
trans_lstm_kernel(const float* __restrict__ vecs,
                  const float* __restrict__ WhhF, const float* __restrict__ WhhB,
                  float* __restrict__ ws) {
  __shared__ float xb[2][16 * 64];   // double-buffered x tile (cols 50..63 = 0)
  __shared__ float z[16 * 128];
  __shared__ float h[16 * 32];
  const int lane = threadIdx.x;
  const int row0 = blockIdx.x * 16;
  float* prog = ws + WS_PROG;

  for (int i = lane; i < 1024; i += 32) { xb[0][i] = 0.0f; xb[1][i] = 0.0f; }

  auto stage = [&](int t, int buf) {   // 16 rows x 50 floats = exactly 25 waves of 32
#pragma unroll
    for (int s = 0; s < 25; ++s) {
      int p = lane + 32 * s;
      int m = p / 50, k = p % 50;
      cp_b32(vecs + (size_t)(row0 + m) * (TTRANS * MAXTAGS) + t * MAXTAGS + k,
             &xb[buf][m * 64 + k]);
    }
  };

  for (int dir = 0; dir < 2; ++dir) {
    const float* wc   = ws + (dir ? WS_WCB : WS_WCF);
    const float* bias = ws + (dir ? WS_BB  : WS_BF);
    const float* Whh  = dir ? WhhB : WhhF;

    for (int i = lane; i < 512; i += 32) h[i] = 0.0f;
    float creg[16];
#pragma unroll
    for (int q = 0; q < 16; ++q) creg[q] = 0.0f;

    v16h bwc[2][8], bhh[8];
    float bv[8];
#pragma unroll
    for (int nt = 0; nt < 8; ++nt) {
#pragma unroll
      for (int kc = 0; kc < 2; ++kc)
        bwc[kc][nt] = load_B([&](int k, int n) -> float {
          return wc[(nt * 16 + n) * 64 + kc * 32 + k];   // already zero-padded
        }, lane);
      bhh[nt] = load_B([&](int k, int n) -> float {
        return Whh[(nt * 16 + n) * 32 + k];
      }, lane);
      bv[nt] = bias[nt * 16 + (lane & 15)];
    }

    stage(dir ? (TTRANS - 1) : 0, 0);
    async_wait();
    __syncthreads();

    for (int s = 0; s < TTRANS; ++s) {
      const int cur = s & 1;
      if (s + 1 < TTRANS) stage(dir ? (TTRANS - 2 - s) : (s + 1), cur ^ 1);
      v16h ax0 = load_A([&](int m, int k) -> float { return xb[cur][m * 64 + k]; }, lane);
      v16h ax1 = load_A([&](int m, int k) -> float { return xb[cur][m * 64 + 32 + k]; }, lane);
      v16h ah  = load_A([&](int m, int k) -> float { return h[m * 32 + k]; }, lane);
#pragma unroll
      for (int nt = 0; nt < 8; ++nt) {
        v8f d = acc_bias(bv[nt]);
        d = wmma16(ax0, bwc[0][nt], d);
        d = wmma16(ax1, bwc[1][nt], d);
        d = wmma16(ah,  bhh[nt],    d);
        store_D(d, lane, [&](int m, int n, float v) { z[m * 128 + nt * 16 + n] = v; });
      }
      __syncthreads();
#pragma unroll
      for (int q = 0; q < 16; ++q) {       // lane owns hidden col j=lane
        float zi = z[q * 128 + lane];
        float zf = z[q * 128 + 32 + lane];
        float zg = z[q * 128 + 64 + lane];
        float zo = z[q * 128 + 96 + lane];
        float c = sigf(zf) * creg[q] + sigf(zi) * tanhf(zg);
        creg[q] = c;
        h[q * 32 + lane] = sigf(zo) * tanhf(c);
      }
      async_wait();
      __syncthreads();
    }
    for (int q = 0; q < 16; ++q)
      prog[(size_t)(row0 + q) * 64 + dir * 32 + lane] = h[q * 32 + lane];
    __syncthreads();
  }
}

// ================= K3: comp MLP 851 -> 128 -> 64 -> 32 -> 16, ELU ============
__global__ void __launch_bounds__(32)
comp_mlp_kernel(const float* __restrict__ fp1, const float* __restrict__ fp3,
                const float* __restrict__ W0, const float* __restrict__ b0,
                const float* __restrict__ W1, const float* __restrict__ b1,
                const float* __restrict__ W2, const float* __restrict__ b2,
                const float* __restrict__ W3, const float* __restrict__ b3,
                float* __restrict__ ws) {
  __shared__ float xb[16 * 864];     // staged concat input rows, padded to 864
  __shared__ float h1[16 * 128];
  __shared__ float h2[16 * 64];
  __shared__ float h3[16 * 32];
  const int lane = threadIdx.x;
  const int row0 = blockIdx.x * 16;
  const float* prog = ws + WS_PROG;
  const float* ex   = ws + WS_EXPR;
  float* cemb = ws + WS_CEMB;

  // ---- stage 16 x 851 rows into LDS via async b128 copies (all segments 16B aligned)
  for (int m = 0; m < 16; ++m) {
    size_t r = (size_t)(row0 + m);
    float* dst = xb + m * 864;
    for (int c = lane; c < 150; c += 32) cp_b128(fp1 + r * 600 + c * 4, dst + c * 4);
    if (lane < 16) cp_b128(prog + r * 64 + lane * 4, dst + 600 + lane * 4);
    for (int c = lane; c < 44; c += 32) cp_b128(fp3 + r * 176 + c * 4, dst + 664 + c * 4);
    if (lane < 4) cp_b128(ex + r * 16 + lane * 4, dst + 840 + lane * 4);  // cols 851..855 = 0 (expr pad)
    if (lane < 8) dst[856 + lane] = 0.0f;
  }
  async_wait();
  __syncthreads();

  // ---- layer 1: K=864 (27 chunks), N=128
  v8f acc[8];
#pragma unroll
  for (int nt = 0; nt < 8; ++nt) acc[nt] = acc_bias(b0[nt * 16 + (lane & 15)]);
  for (int kc = 0; kc < 27; ++kc) {
    v16h a = load_A([&](int m, int k) -> float { return xb[m * 864 + kc * 32 + k]; }, lane);
#pragma unroll
    for (int nt = 0; nt < 8; ++nt) {
      v16h b = load_B([&](int k, int n) -> float {
        int kg = kc * 32 + k;
        return (kg < 851) ? W0[(size_t)(nt * 16 + n) * 851 + kg] : 0.0f;
      }, lane);
      acc[nt] = wmma16(a, b, acc[nt]);
    }
  }
#pragma unroll
  for (int nt = 0; nt < 8; ++nt)
    store_D(acc[nt], lane, [&](int m, int n, float v) { h1[m * 128 + nt * 16 + n] = eluf(v); });
  __syncthreads();

  // ---- layer 2: 128 -> 64
#pragma unroll
  for (int nt = 0; nt < 4; ++nt) {
    v8f d = acc_bias(b1[nt * 16 + (lane & 15)]);
#pragma unroll
    for (int kc = 0; kc < 4; ++kc) {
      v16h a = load_A([&](int m, int k) -> float { return h1[m * 128 + kc * 32 + k]; }, lane);
      v16h b = load_B([&](int k, int n) -> float { return W1[(nt * 16 + n) * 128 + kc * 32 + k]; }, lane);
      d = wmma16(a, b, d);
    }
    store_D(d, lane, [&](int m, int n, float v) { h2[m * 64 + nt * 16 + n] = eluf(v); });
  }
  __syncthreads();

  // ---- layer 3: 64 -> 32
#pragma unroll
  for (int nt = 0; nt < 2; ++nt) {
    v8f d = acc_bias(b2[nt * 16 + (lane & 15)]);
#pragma unroll
    for (int kc = 0; kc < 2; ++kc) {
      v16h a = load_A([&](int m, int k) -> float { return h2[m * 64 + kc * 32 + k]; }, lane);
      v16h b = load_B([&](int k, int n) -> float { return W2[(nt * 16 + n) * 64 + kc * 32 + k]; }, lane);
      d = wmma16(a, b, d);
    }
    store_D(d, lane, [&](int m, int n, float v) { h3[m * 32 + nt * 16 + n] = eluf(v); });
  }
  __syncthreads();

  // ---- layer 4: 32 -> 16, write comps_emb
  {
    v8f d = acc_bias(b3[lane & 15]);
    v16h a = load_A([&](int m, int k) -> float { return h3[m * 32 + k]; }, lane);
    v16h b = load_B([&](int k, int n) -> float { return W3[n * 32 + k]; }, lane);
    d = wmma16(a, b, d);
    store_D(d, lane, [&](int m, int n, float v) { cemb[(size_t)(row0 + m) * 16 + n] = eluf(v); });
  }
}

// ================= K4: tree recursion + head (batch 256, scalar) =============
struct LSTMP { const float *Wih, *Whh, *b; };   // b = bih + bhh (pre-summed)

__device__ void lstm16_seq(const LSTMP& p, const float* xs, int stride, int steps, float* hout) {
  float h[16], c[16];
#pragma unroll
  for (int j = 0; j < 16; ++j) { h[j] = 0.0f; c[j] = 0.0f; }
  for (int s = 0; s < steps; ++s) {
    const float* x = xs + s * stride;
    float nh[16];
#pragma unroll
    for (int j = 0; j < 16; ++j) {
      float zi = p.b[j];
      float zf = p.b[16 + j];
      float zg = p.b[32 + j];
      float zo = p.b[48 + j];
      for (int k = 0; k < 16; ++k) {
        float xv = x[k], hv = h[k];
        zi += xv * p.Wih[j * 16 + k]        + hv * p.Whh[j * 16 + k];
        zf += xv * p.Wih[(16 + j) * 16 + k] + hv * p.Whh[(16 + j) * 16 + k];
        zg += xv * p.Wih[(32 + j) * 16 + k] + hv * p.Whh[(32 + j) * 16 + k];
        zo += xv * p.Wih[(48 + j) * 16 + k] + hv * p.Whh[(48 + j) * 16 + k];
      }
      float cv = sigf(zf) * c[j] + sigf(zi) * tanhf(zg);
      c[j]  = cv;
      nh[j] = sigf(zo) * tanhf(cv);
    }
#pragma unroll
    for (int j = 0; j < 16; ++j) h[j] = nh[j];
  }
#pragma unroll
  for (int j = 0; j < 16; ++j) hout[j] = h[j];
}

__device__ void concat_mlp(const float* nodes_h, const float* comps_h, const float* loop,
                           const float* W0, const float* b0,
                           const float* W1, const float* b1, float* out) {
  float x[40];
#pragma unroll
  for (int j = 0; j < 16; ++j) x[j] = nodes_h[j];
#pragma unroll
  for (int j = 0; j < 16; ++j) x[16 + j] = comps_h[j];
#pragma unroll
  for (int j = 0; j < 8; ++j)  x[32 + j] = loop[j];
  float t[32];
  for (int o = 0; o < 32; ++o) {
    float s = b0[o];
    for (int k = 0; k < 40; ++k) s += x[k] * W0[o * 40 + k];
    t[o] = eluf(s);
  }
  for (int o = 0; o < 16; ++o) {
    float s = b1[o];
    for (int k = 0; k < 32; ++k) s += t[k] * W1[o * 32 + k];
    out[o] = eluf(s);
  }
}

// LDS weight-cache layout for the tree kernel
enum {
  O_CW0 = 0,            O_CB0 = O_CW0 + 1280, O_CW1 = O_CB0 + 32,  O_CB1 = O_CW1 + 512,
  O_RW0 = O_CB1 + 16,   O_RB0 = O_RW0 + 512,  O_RW1 = O_RB0 + 32,  O_RB1 = O_RW1 + 512,
  O_PW  = O_RB1 + 16,   O_PB  = O_PW + 16,    O_NOC = O_PB + 1,    O_NON = O_NOC + 16,
  O_COMW = O_NON + 16,  O_COMU = O_COMW + 1024, O_COMB = O_COMU + 1024,
  O_NODW = O_COMB + 64, O_NODU = O_NODW + 1024, O_NODB = O_NODU + 1024,
  O_ROOW = O_NODB + 64, O_ROOU = O_ROOW + 1024, O_ROOB = O_ROOU + 1024,
  O_WEND = O_ROOB + 64
};

__global__ void __launch_bounds__(64)
tree_kernel(const float* __restrict__ loops, float* __restrict__ out,
            const float* __restrict__ ws,
            const float* cW0, const float* cb0, const float* cW1, const float* cb1,
            const float* rW0, const float* rb0, const float* rW1, const float* rb1,
            const float* pW,  const float* pb,
            const float* no_comps, const float* no_nodes,
            const float* comWih, const float* comWhh, const float* comBih, const float* comBhh,
            const float* nodWih, const float* nodWhh, const float* nodBih, const float* nodBhh,
            const float* rooWih, const float* rooWhh, const float* rooBih, const float* rooBhh) {
  __shared__ float wb[O_WEND];
  const int tid = threadIdx.x;

  for (int i = tid; i < 1280; i += 64) wb[O_CW0 + i] = cW0[i];
  for (int i = tid; i < 512;  i += 64) { wb[O_CW1 + i] = cW1[i]; wb[O_RW0 + i] = rW0[i]; wb[O_RW1 + i] = rW1[i]; }
  for (int i = tid; i < 32;   i += 64) { wb[O_CB0 + i] = cb0[i]; wb[O_RB0 + i] = rb0[i]; }
  for (int i = tid; i < 16;   i += 64) {
    wb[O_CB1 + i] = cb1[i]; wb[O_RB1 + i] = rb1[i]; wb[O_PW + i] = pW[i];
    wb[O_NOC + i] = no_comps[i]; wb[O_NON + i] = no_nodes[i];
  }
  if (tid == 0) wb[O_PB] = pb[0];
  for (int i = tid; i < 1024; i += 64) {
    wb[O_COMW + i] = comWih[i]; wb[O_COMU + i] = comWhh[i];
    wb[O_NODW + i] = nodWih[i]; wb[O_NODU + i] = nodWhh[i];
    wb[O_ROOW + i] = rooWih[i]; wb[O_ROOU + i] = rooWhh[i];
  }
  for (int i = tid; i < 64; i += 64) {
    wb[O_COMB + i] = comBih[i] + comBhh[i];
    wb[O_NODB + i] = nodBih[i] + nodBhh[i];
    wb[O_ROOB + i] = rooBih[i] + rooBhh[i];
  }
  __syncthreads();

  const int b = blockIdx.x * blockDim.x + tid;
  if (b >= BB) return;
  const float* cemb = ws + WS_CEMB + (size_t)b * CC * 16;
  const float* lp   = loops + (size_t)b * NLOOPS * LOOPSF;
  LSTMP compsL{&wb[O_COMW], &wb[O_COMU], &wb[O_COMB]};
  LSTMP nodesL{&wb[O_NODW], &wb[O_NODU], &wb[O_NODB]};
  LSTMP rootsL{&wb[O_ROOW], &wb[O_ROOU], &wb[O_ROOB]};
  const float* cw0 = &wb[O_CW0]; const float* cb0l = &wb[O_CB0];
  const float* cw1 = &wb[O_CW1]; const float* cb1l = &wb[O_CB1];

  float ch[16], nh[16], leaf1[16], leaf2[16], leaf4[16], s0[16], s1[16];

  // leaf(loop1, comps 0..7)
  lstm16_seq(compsL, cemb + 0 * 16, 16, 8, ch);
  concat_mlp(&wb[O_NON], ch, lp + 1 * LOOPSF, cw0, cb0l, cw1, cb1l, leaf1);
  // leaf(loop2, comps 8..15)
  lstm16_seq(compsL, cemb + 8 * 16, 16, 8, ch);
  concat_mlp(&wb[O_NON], ch, lp + 2 * LOOPSF, cw0, cb0l, cw1, cb1l, leaf2);
  // root0 (loop0): nodes_lstm over [leaf1, leaf2], no comps
  {
    float seq[32];
    for (int j = 0; j < 16; ++j) { seq[j] = leaf1[j]; seq[16 + j] = leaf2[j]; }
    lstm16_seq(nodesL, seq, 16, 2, nh);
    concat_mlp(nh, &wb[O_NOC], lp + 0 * LOOPSF, cw0, cb0l, cw1, cb1l, s0);
  }
  // leaf(loop4, comps 24..31)
  lstm16_seq(compsL, cemb + 24 * 16, 16, 8, ch);
  concat_mlp(&wb[O_NON], ch, lp + 4 * LOOPSF, cw0, cb0l, cw1, cb1l, leaf4);
  // root1 (loop3): nodes_lstm over [leaf4], comps 16..23
  lstm16_seq(nodesL, leaf4, 16, 1, nh);
  lstm16_seq(compsL, cemb + 16 * 16, 16, 8, ch);
  concat_mlp(nh, ch, lp + 3 * LOOPSF, cw0, cb0l, cw1, cb1l, s1);

  // roots_lstm over [s0, s1]
  float rh[16];
  {
    float seq[32];
    for (int j = 0; j < 16; ++j) { seq[j] = s0[j]; seq[16 + j] = s1[j]; }
    lstm16_seq(rootsL, seq, 16, 2, rh);
  }
  // reg MLP 16 -> 32 -> 16 (ELU), predict, leaky output
  float t[32], y[16];
  for (int o = 0; o < 32; ++o) {
    float s = wb[O_RB0 + o];
    for (int k = 0; k < 16; ++k) s += rh[k] * wb[O_RW0 + o * 16 + k];
    t[o] = eluf(s);
  }
  for (int o = 0; o < 16; ++o) {
    float s = wb[O_RB1 + o];
    for (int k = 0; k < 32; ++k) s += t[k] * wb[O_RW1 + o * 32 + k];
    y[o] = eluf(s);
  }
  float p = wb[O_PB];
  for (int k = 0; k < 16; ++k) p += y[k] * wb[O_PW + k];
  out[b] = p > 0.0f ? p : 0.01f * p;
}

// ================= launch =================
extern "C" void kernel_launch(void* const* d_in, const int* in_sizes, int n_in,
                              void* d_out, int out_size, void* d_ws, size_t ws_size,
                              hipStream_t stream) {
  (void)in_sizes; (void)n_in; (void)out_size; (void)ws_size;
  const float* expr_tree = (const float*)d_in[0];
  const float* fp1       = (const float*)d_in[1];
  const float* vecs      = (const float*)d_in[2];
  const float* fp3       = (const float*)d_in[3];
  const float* loops     = (const float*)d_in[4];
  const float* encW = (const float*)d_in[5];  const float* encB = (const float*)d_in[6];
  const float* c0W  = (const float*)d_in[7];  const float* c0B  = (const float*)d_in[8];
  const float* c1W  = (const float*)d_in[9];  const float* c1B  = (const float*)d_in[10];
  const float* c2W  = (const float*)d_in[11]; const float* c2B  = (const float*)d_in[12];
  const float* c3W  = (const float*)d_in[13]; const float* c3B  = (const float*)d_in[14];
  const float* cc0W = (const float*)d_in[15]; const float* cc0B = (const float*)d_in[16];
  const float* cc1W = (const float*)d_in[17]; const float* cc1B = (const float*)d_in[18];
  const float* rg0W = (const float*)d_in[19]; const float* rg0B = (const float*)d_in[20];
  const float* rg1W = (const float*)d_in[21]; const float* rg1B = (const float*)d_in[22];
  const float* pW   = (const float*)d_in[23]; const float* pB   = (const float*)d_in[24];
  const float* noC  = (const float*)d_in[25]; const float* noN  = (const float*)d_in[26];
  const float* comWih = (const float*)d_in[27]; const float* comWhh = (const float*)d_in[28];
  const float* comBih = (const float*)d_in[29]; const float* comBhh = (const float*)d_in[30];
  const float* nodWih = (const float*)d_in[31]; const float* nodWhh = (const float*)d_in[32];
  const float* nodBih = (const float*)d_in[33]; const float* nodBhh = (const float*)d_in[34];
  const float* rooWih = (const float*)d_in[35]; const float* rooWhh = (const float*)d_in[36];
  const float* rooBih = (const float*)d_in[37]; const float* rooBhh = (const float*)d_in[38];
  const float* tfWih  = (const float*)d_in[39]; const float* tfWhh  = (const float*)d_in[40];
  const float* tfBih  = (const float*)d_in[41]; const float* tfBhh  = (const float*)d_in[42];
  const float* tbWih  = (const float*)d_in[43]; const float* tbWhh  = (const float*)d_in[44];
  const float* tbBih  = (const float*)d_in[45]; const float* tbBhh  = (const float*)d_in[46];
  const float* exWih  = (const float*)d_in[47]; const float* exWhh  = (const float*)d_in[48];
  const float* exBih  = (const float*)d_in[49]; const float* exBhh  = (const float*)d_in[50];

  float* ws  = (float*)d_ws;
  float* out = (float*)d_out;

  prep_kernel<<<1, 256, 0, stream>>>(encW, encB, tfWih, tfBih, tfBhh, tbWih, tbBih, tbBhh, ws);
  expr_lstm_kernel<<<BC / 16, 32, 0, stream>>>(expr_tree, exWih, exWhh, exBih, exBhh, ws);
  trans_lstm_kernel<<<BC / 16, 32, 0, stream>>>(vecs, tfWhh, tbWhh, ws);
  comp_mlp_kernel<<<BC / 16, 32, 0, stream>>>(fp1, fp3, c0W, c0B, c1W, c1B, c2W, c2B, c3W, c3B, ws);
  tree_kernel<<<(BB + 63) / 64, 64, 0, stream>>>(loops, out, ws,
      cc0W, cc0B, cc1W, cc1B, rg0W, rg0B, rg1W, rg1B, pW, pB, noC, noN,
      comWih, comWhh, comBih, comBhh, nodWih, nodWhh, nodBih, nodBhh,
      rooWih, rooWhh, rooBih, rooBhh);
}
